// GraphNeuralOperator_66194035965973
// MI455X (gfx1250) — matrix operand order repaired and checked by reading, not compiled, not run on hardware
//
#include <hip/hip_runtime.h>

// ---------------- problem constants (from reference) ----------------
#define BATCH 2
#define NNODE 20000
#define NEDGE 320000
#define NDIM  64
#define EDIM  32
#define HDIM  256
#define NLAY  4
#define MIN_K (2 * HDIM + EDIM)   // 544

// ---------------- WMMA types ----------------
typedef __attribute__((ext_vector_type(16))) __bf16 v16bf;
typedef __attribute__((ext_vector_type(8)))  __bf16 v8bf;
typedef __attribute__((ext_vector_type(8)))  float  v8f;
typedef __attribute__((ext_vector_type(4)))  float  v4f;
typedef int v4i __attribute__((vector_size(16)));   // matches builtin param type

static __device__ __forceinline__ v8f wmma_bf16(v16bf a, v16bf b, v8f c) {
  // D = A(16x32 bf16) * B(32x16 bf16) + C(16x16 f32)
  return __builtin_amdgcn_wmma_f32_16x16x32_bf16(false, a, false, b, (short)0, c,
                                                 false, false);
}

// ---------------- async global -> LDS copy (CDNA5 ASYNCcnt path) -----------
static __device__ __forceinline__ void async_b128(const float* g, float* lds) {
#if __has_builtin(__builtin_amdgcn_global_load_async_to_lds_b128)
  __builtin_amdgcn_global_load_async_to_lds_b128(
      (__attribute__((address_space(1))) v4i*)(v4i*)(void*)(float*)g,
      (__attribute__((address_space(3))) v4i*)(v4i*)(void*)lds,
      0, 0);
#else
  unsigned l = (unsigned)(unsigned long long)lds;   // LDS offset = low 32 bits
  asm volatile("global_load_async_to_lds_b128 %0, %1, off"
               :: "v"(l), "v"((unsigned long long)g) : "memory");
#endif
}

static __device__ __forceinline__ void wait_async0() {
#if __has_builtin(__builtin_amdgcn_s_wait_asynccnt)
  __builtin_amdgcn_s_wait_asynccnt(0);
#else
  asm volatile("s_wait_asynccnt 0x0" ::: "memory");
#endif
}

// ---------------- fragment loaders -----------------------------------------
// A-fragment from an f32 LDS tile with on-the-fly bf16 convert.
// ISA 16-bit A layout: lane m=L%16 is row M; lanes 0-15 hold K in
// {0..7, 16..23}+k_base, lanes 16-31 hold {8..15, 24..31}+k_base.
// cvt VALU co-executes with WMMA (XDL 16-bit ops are TRANS-tracked).
static __device__ __forceinline__ v16bf lds_a_frag_f32(const float* tile,
                                                       int row_stride, int k_base,
                                                       int lane) {
  const float* p = tile + (lane & 15) * row_stride + k_base + ((lane >> 4) << 3);
  v4f x0 = *(const v4f*)(p);
  v4f x1 = *(const v4f*)(p + 4);
  v4f x2 = *(const v4f*)(p + 16);
  v4f x3 = *(const v4f*)(p + 20);
  v16bf r;
#pragma unroll
  for (int j = 0; j < 4; ++j) {
    r[j]      = (__bf16)x0[j];
    r[4 + j]  = (__bf16)x1[j];
    r[8 + j]  = (__bf16)x2[j];
    r[12 + j] = (__bf16)x3[j];
  }
  return r;
}

// A-fragment from a bf16 LDS tile (two 16-byte ds_load_b128 per lane).
static __device__ __forceinline__ v16bf lds_a_frag(const __bf16* tile,
                                                   int row_stride, int k_base,
                                                   int lane) {
  const __bf16* p = tile + (lane & 15) * row_stride + k_base + ((lane >> 4) << 3);
  union { v16bf v; v8bf h[2]; } u;
  u.h[0] = *(const v8bf*)(p);
  u.h[1] = *(const v8bf*)(p + 16);
  return u.v;
}

// B-fragment from pre-packed global weights: each lane reads 16 contiguous bf16.
static __device__ __forceinline__ v16bf gbl_b_frag(const __bf16* __restrict__ packed,
                                                   int blk, int lane) {
  return *(const v16bf*)(packed + ((long)blk << 9) + (lane << 4));
}

static __device__ __forceinline__ float gelu_tanh(float x) {
  const float k0 = 0.7978845608028654f;   // sqrt(2/pi)
  const float k1 = 0.044715f;
  float t = tanhf(k0 * (x + k1 * x * x * x));
  return 0.5f * x * (1.0f + t);
}

// ---------------- weight packing: f32 row-major (K,N) -> bf16 B-fragments ----
// dst[((mat*Ksteps + ks)*Ntiles + nt)*512 + lane*16 + j]
//   = W[mat][ks*32 + (j/8)*16 + (lane/16)*8 + j%8][nt*16 + lane%16]
__global__ void pack_b(const float* __restrict__ src, __bf16* __restrict__ dst,
                       int K, int Nc, int nmat) {
  const long per = (long)K * Nc;
  const long total = per * nmat;
  const int ntiles = Nc >> 4;
  for (long i = (long)blockIdx.x * blockDim.x + threadIdx.x; i < total;
       i += (long)gridDim.x * blockDim.x) {
    int  mat = (int)(i / per);
    long r   = i - (long)mat * per;
    int  j    = (int)(r & 15);
    int  lane = (int)((r >> 4) & 31);
    long blk  = r >> 9;
    int  ks   = (int)(blk / ntiles);
    int  nt   = (int)(blk - (long)ks * ntiles);
    int  kk   = ks * 32 + ((j >> 3) << 4) + ((lane >> 4) << 3) + (j & 7);
    int  nn   = (nt << 4) + (lane & 15);
    dst[i] = (__bf16)src[(long)mat * per + (long)kk * Nc + nn];
  }
}

__global__ void zero_f32(float* __restrict__ p, long n) {
  for (long i = (long)blockIdx.x * blockDim.x + threadIdx.x; i < n;
       i += (long)gridDim.x * blockDim.x) p[i] = 0.0f;
}

// ---------------- input projection: h = nf(.,64) @ inW(64,256) + b ----------
__global__ void __launch_bounds__(256)
input_proj(const float* __restrict__ nf, const __bf16* __restrict__ Wp,
           const float* __restrict__ bias, float* __restrict__ h) {
  __shared__ alignas(16) float A[16 * NDIM];
  const int tid = threadIdx.x;
  const long row0 = (long)blockIdx.x * 16;
  // tile rows are contiguous in nf: one async b128 per thread
  for (int i = tid; i < 16 * NDIM / 4; i += 256)
    async_b128(&nf[row0 * NDIM + i * 4], &A[i * 4]);
  wait_async0();
  __syncthreads();

  const int wave = tid >> 5, lane = tid & 31;
  const int nt0 = wave * 2, nt1 = nt0 + 1;
  v8f a0 = {}, a1 = {};
  for (int ks = 0; ks < NDIM / 32; ++ks) {
    v16bf a  = lds_a_frag_f32(A, NDIM, ks * 32, lane);
    v16bf b0 = gbl_b_frag(Wp, ks * 16 + nt0, lane);
    v16bf b1 = gbl_b_frag(Wp, ks * 16 + nt1, lane);
    a0 = wmma_bf16(a, b0, a0);
    a1 = wmma_bf16(a, b1, a1);
  }
  const int n0 = nt0 * 16 + (lane & 15), n1 = nt1 * 16 + (lane & 15);
  const float bb0 = bias[n0], bb1 = bias[n1];
  const int mb = (lane >> 4) * 8;
  for (int r = 0; r < 8; ++r) {
    h[(row0 + mb + r) * HDIM + n0] = a0[r] + bb0;
    h[(row0 + mb + r) * HDIM + n1] = a1[r] + bb1;
  }
}

// ---------------- edge MLP + scatter-add -----------------------------------
// msg = gelu([h[src],h[dst],ef] @ W1 + b1) @ W2 + b2 ; agg[dst] += msg
__global__ void __launch_bounds__(256)
edge_mlp(const float* __restrict__ h, const int* __restrict__ edges,
         const float* __restrict__ ef,
         const __bf16* __restrict__ W1p, const float* __restrict__ b1,
         const __bf16* __restrict__ W2p, const float* __restrict__ b2,
         float* __restrict__ agg, int layer) {
  __shared__ alignas(16) float  MinF[16 * MIN_K];   // f32 staging via async DMA
  __shared__ alignas(32) __bf16 Hid[16 * HDIM];
  __shared__ int sIdx[16], dIdx[16];

  const int tid = threadIdx.x;
  const int tilesPerBatch = NEDGE / 16;
  const int b  = blockIdx.x / tilesPerBatch;
  const int e0 = (blockIdx.x - b * tilesPerBatch) * 16;
  const long ebase = (long)b * NEDGE + e0;
  const long hbase = (long)b * NNODE * HDIM;

  if (tid < 16) {
    sIdx[tid] = edges[(ebase + tid) * 2 + 0];
    dIdx[tid] = edges[(ebase + tid) * 2 + 1];
  }
  __syncthreads();

  // Gather msg_in rows into LDS with the async copy engine (per-lane scattered
  // 16B transfers, tracked by ASYNCcnt; no VGPR round-trip).
  for (int i = tid; i < 16 * (MIN_K / 4); i += 256) {
    int m  = i / (MIN_K / 4);
    int c  = (i - m * (MIN_K / 4)) * 4;
    const float* g;
    if (c < HDIM)            g = &h[hbase + (long)sIdx[m] * HDIM + c];
    else if (c < 2 * HDIM)   g = &h[hbase + (long)dIdx[m] * HDIM + (c - HDIM)];
    else                     g = &ef[(ebase + m) * EDIM + (c - 2 * HDIM)];
    async_b128(g, &MinF[m * MIN_K + c]);
  }
  wait_async0();
  __syncthreads();

  const int wave = tid >> 5, lane = tid & 31;
  const int nt0 = wave * 2, nt1 = nt0 + 1;
  const int n0 = nt0 * 16 + (lane & 15), n1 = nt1 * 16 + (lane & 15);
  const int mb = (lane >> 4) * 8;

  // GEMM1: (16x544)@(544x256), 17 k-steps
  v8f a0 = {}, a1 = {};
  for (int ks = 0; ks < MIN_K / 32; ++ks) {
    v16bf a  = lds_a_frag_f32(MinF, MIN_K, ks * 32, lane);
    v16bf b0 = gbl_b_frag(W1p, (layer * (MIN_K / 32) + ks) * 16 + nt0, lane);
    v16bf b1 = gbl_b_frag(W1p, (layer * (MIN_K / 32) + ks) * 16 + nt1, lane);
    a0 = wmma_bf16(a, b0, a0);
    a1 = wmma_bf16(a, b1, a1);
  }
  {
    const float bb0 = b1[layer * HDIM + n0], bb1 = b1[layer * HDIM + n1];
    for (int r = 0; r < 8; ++r) {
      Hid[(mb + r) * HDIM + n0] = (__bf16)gelu_tanh(a0[r] + bb0);
      Hid[(mb + r) * HDIM + n1] = (__bf16)gelu_tanh(a1[r] + bb1);
    }
  }
  __syncthreads();

  // GEMM2: (16x256)@(256x256), 8 k-steps
  v8f c0 = {}, c1 = {};
  for (int ks = 0; ks < HDIM / 32; ++ks) {
    v16bf a  = lds_a_frag(Hid, HDIM, ks * 32, lane);
    v16bf b0 = gbl_b_frag(W2p, (layer * (HDIM / 32) + ks) * 16 + nt0, lane);
    v16bf b1 = gbl_b_frag(W2p, (layer * (HDIM / 32) + ks) * 16 + nt1, lane);
    c0 = wmma_bf16(a, b0, c0);
    c1 = wmma_bf16(a, b1, c1);
  }
  {
    const float bb0 = b2[layer * HDIM + n0], bb1 = b2[layer * HDIM + n1];
    for (int r = 0; r < 8; ++r) {
      long row = hbase + (long)dIdx[mb + r] * HDIM;
      atomicAdd(&agg[row + n0], c0[r] + bb0);   // L2-resident f32 atomic
      atomicAdd(&agg[row + n1], c1[r] + bb1);
    }
  }
}

// ---------------- node update MLP + residual -------------------------------
// h += gelu([h,agg] @ V1 + c1) @ V2 + c2
__global__ void __launch_bounds__(256)
node_mlp(float* __restrict__ h, const float* __restrict__ agg,
         const __bf16* __restrict__ V1p, const float* __restrict__ c1,
         const __bf16* __restrict__ V2p, const float* __restrict__ c2, int layer) {
  __shared__ alignas(16) float  UinF[16 * 2 * HDIM];
  __shared__ alignas(32) __bf16 Hid[16 * HDIM];
  const int tid = threadIdx.x;
  const long row0 = (long)blockIdx.x * 16;   // flat over B*N

  for (int i = tid; i < 16 * (2 * HDIM / 4); i += 256) {
    int m = i >> 7, c = (i & 127) * 4;
    const float* g = (c < HDIM) ? &h[(row0 + m) * HDIM + c]
                                : &agg[(row0 + m) * HDIM + (c - HDIM)];
    async_b128(g, &UinF[m * 2 * HDIM + c]);
  }
  wait_async0();
  __syncthreads();

  const int wave = tid >> 5, lane = tid & 31;
  const int nt0 = wave * 2, nt1 = nt0 + 1;
  const int n0 = nt0 * 16 + (lane & 15), n1 = nt1 * 16 + (lane & 15);
  const int mb = (lane >> 4) * 8;

  v8f a0 = {}, a1 = {};
  for (int ks = 0; ks < (2 * HDIM) / 32; ++ks) {        // 16 k-steps
    v16bf a  = lds_a_frag_f32(UinF, 2 * HDIM, ks * 32, lane);
    v16bf b0 = gbl_b_frag(V1p, (layer * 16 + ks) * 16 + nt0, lane);
    v16bf b1 = gbl_b_frag(V1p, (layer * 16 + ks) * 16 + nt1, lane);
    a0 = wmma_bf16(a, b0, a0);
    a1 = wmma_bf16(a, b1, a1);
  }
  {
    const float bb0 = c1[layer * HDIM + n0], bb1 = c1[layer * HDIM + n1];
    for (int r = 0; r < 8; ++r) {
      Hid[(mb + r) * HDIM + n0] = (__bf16)gelu_tanh(a0[r] + bb0);
      Hid[(mb + r) * HDIM + n1] = (__bf16)gelu_tanh(a1[r] + bb1);
    }
  }
  __syncthreads();

  v8f d0 = {}, d1 = {};
  for (int ks = 0; ks < HDIM / 32; ++ks) {               // 8 k-steps
    v16bf a  = lds_a_frag(Hid, HDIM, ks * 32, lane);
    v16bf b0 = gbl_b_frag(V2p, (layer * 8 + ks) * 16 + nt0, lane);
    v16bf b1 = gbl_b_frag(V2p, (layer * 8 + ks) * 16 + nt1, lane);
    d0 = wmma_bf16(a, b0, d0);
    d1 = wmma_bf16(a, b1, d1);
  }
  {
    const float bb0 = c2[layer * HDIM + n0], bb1 = c2[layer * HDIM + n1];
    for (int r = 0; r < 8; ++r) {
      long i0 = (row0 + mb + r) * HDIM + n0;
      long i1 = (row0 + mb + r) * HDIM + n1;
      h[i0] = h[i0] + d0[r] + bb0;   // residual; unique writer per element
      h[i1] = h[i1] + d1[r] + bb1;
    }
  }
}

// ---------------- output projection: out = h @ outW(256,64) + b ------------
__global__ void __launch_bounds__(128)
out_proj(const float* __restrict__ h, const __bf16* __restrict__ Wp,
         const float* __restrict__ bias, float* __restrict__ out) {
  __shared__ alignas(16) float A[16 * HDIM];
  const int tid = threadIdx.x;
  const long row0 = (long)blockIdx.x * 16;
  for (int i = tid; i < 16 * HDIM / 4; i += 128)
    async_b128(&h[row0 * HDIM + i * 4], &A[i * 4]);
  wait_async0();
  __syncthreads();

  const int wave = tid >> 5, lane = tid & 31;
  const int nt = wave;                        // 4 N-tiles of 16 (N=64)
  v8f acc = {};
  for (int ks = 0; ks < HDIM / 32; ++ks) {
    v16bf a = lds_a_frag_f32(A, HDIM, ks * 32, lane);
    v16bf b = gbl_b_frag(Wp, ks * 4 + nt, lane);
    acc = wmma_bf16(a, b, acc);
  }
  const int n = nt * 16 + (lane & 15);
  const float bb = bias[n];
  const int mb = (lane >> 4) * 8;
  for (int r = 0; r < 8; ++r) out[(row0 + mb + r) * NDIM + n] = acc[r] + bb;
}

// ---------------- host orchestration ---------------------------------------
extern "C" void kernel_launch(void* const* d_in, const int* in_sizes, int n_in,
                              void* d_out, int out_size, void* d_ws, size_t ws_size,
                              hipStream_t stream) {
  const float* nf   = (const float*)d_in[0];
  const int*   ei   = (const int*)  d_in[1];   // edge_indices (int32 on device)
  const float* ef   = (const float*)d_in[2];
  const float* inW  = (const float*)d_in[3];
  const float* inb  = (const float*)d_in[4];
  const float* mW1  = (const float*)d_in[5];
  const float* mb1  = (const float*)d_in[6];
  const float* mW2  = (const float*)d_in[7];
  const float* mb2  = (const float*)d_in[8];
  const float* uW1  = (const float*)d_in[9];
  const float* ub1  = (const float*)d_in[10];
  const float* uW2  = (const float*)d_in[11];
  const float* ub2  = (const float*)d_in[12];
  const float* outW = (const float*)d_in[13];
  const float* outb = (const float*)d_in[14];
  float* out = (float*)d_out;

  char* ws = (char*)d_ws;
  size_t off = 0;
  auto alloc = [&](size_t bytes) -> char* {
    off = (off + 255) & ~(size_t)255;
    char* p = ws + off;
    off += bytes;
    return p;
  };

  float*  h     = (float*) alloc((size_t)BATCH * NNODE * HDIM * 4);
  float*  agg   = (float*) alloc((size_t)BATCH * NNODE * HDIM * 4);
  __bf16* inWp  = (__bf16*)alloc((size_t)NDIM * HDIM * 2);
  __bf16* mW1p  = (__bf16*)alloc((size_t)NLAY * MIN_K * HDIM * 2);
  __bf16* mW2p  = (__bf16*)alloc((size_t)NLAY * HDIM * HDIM * 2);
  __bf16* uW1p  = (__bf16*)alloc((size_t)NLAY * 2 * HDIM * HDIM * 2);
  __bf16* uW2p  = (__bf16*)alloc((size_t)NLAY * HDIM * HDIM * 2);
  __bf16* outWp = (__bf16*)alloc((size_t)HDIM * NDIM * 2);

  pack_b<<<64,  256, 0, stream>>>(inW,  inWp,  NDIM,     HDIM, 1);
  pack_b<<<256, 256, 0, stream>>>(mW1,  mW1p,  MIN_K,    HDIM, NLAY);
  pack_b<<<256, 256, 0, stream>>>(mW2,  mW2p,  HDIM,     HDIM, NLAY);
  pack_b<<<256, 256, 0, stream>>>(uW1,  uW1p,  2 * HDIM, HDIM, NLAY);
  pack_b<<<256, 256, 0, stream>>>(uW2,  uW2p,  HDIM,     HDIM, NLAY);
  pack_b<<<64,  256, 0, stream>>>(outW, outWp, HDIM,     NDIM, 1);

  input_proj<<<BATCH * NNODE / 16, 256, 0, stream>>>(nf, inWp, inb, h);

  for (int l = 0; l < NLAY; ++l) {
    zero_f32<<<2048, 256, 0, stream>>>(agg, (long)BATCH * NNODE * HDIM);
    edge_mlp<<<BATCH * NEDGE / 16, 256, 0, stream>>>(h, ei, ef, mW1p, mb1,
                                                     mW2p, mb2, agg, l);
    node_mlp<<<BATCH * NNODE / 16, 256, 0, stream>>>(h, agg, uW1p, ub1,
                                                     uW2p, ub2, l);
  }

  out_proj<<<BATCH * NNODE / 16, 128, 0, stream>>>(h, outWp, outb, out);
}
